// GlobalLocalAttention_43825846288470
// MI455X (gfx1250) — compile-verified, matrix operand order checked by reading, not run on hardware
//
#include <hip/hip_runtime.h>
#include <hip/hip_bf16.h>

typedef __attribute__((ext_vector_type(16))) __bf16 v16bf;
typedef __attribute__((ext_vector_type(8)))  __bf16 v8bf;
typedef __attribute__((ext_vector_type(8)))  float  v8f;
typedef __attribute__((ext_vector_type(4)))  float  f32x4;

#define S_LEN  2048
#define DMODEL 1024
#define NHEAD  16
#define DH     64
#define WIN    256

union FragAB { v16bf v; v8bf h[2]; };

// Build a 16x32(K) bf16 WMMA fragment: per-lane two contiguous 16B pieces.
static __device__ inline v16bf make_frag(const __bf16* lo, const __bf16* hi) {
  FragAB f;
  f.h[0] = *(const v8bf*)lo;
  f.h[1] = *(const v8bf*)hi;
  return f.v;
}

// ---------------------------------------------------------------------------
// CDNA5 async memory->LDS copy (16B per lane), tracked by ASYNCcnt.
// ---------------------------------------------------------------------------
static __device__ inline void async_copy_b128(void* lds_dst, const void* gsrc) {
  unsigned loff = (unsigned)(size_t)lds_dst;  // low 32 bits = LDS byte offset
  asm volatile("global_load_async_to_lds_b128 %0, %1, off"
               :: "v"(loff), "v"(gsrc) : "memory");
}

static __device__ inline void wait_async0() {
#if __has_builtin(__builtin_amdgcn_s_wait_asynccnt)
  __builtin_amdgcn_s_wait_asynccnt(0);
#else
  asm volatile("s_wait_asynccnt 0" ::: "memory");
#endif
}

static __device__ inline void wait_async3() {
#if __has_builtin(__builtin_amdgcn_s_wait_asynccnt)
  __builtin_amdgcn_s_wait_asynccnt(3);
#else
  asm volatile("s_wait_asynccnt 3" ::: "memory");
#endif
}

// ---------------------------------------------------------------------------
// f32 -> bf16 bulk convert (8 elems / thread, b128 loads)
// ---------------------------------------------------------------------------
__global__ __launch_bounds__(256) void cvt_bf16_kernel(const float* __restrict__ src,
                                                       __bf16* __restrict__ dst, int n) {
  int i = (blockIdx.x * blockDim.x + threadIdx.x) * 8;
  if (i >= n) return;
  f32x4 a = *(const f32x4*)(src + i);
  f32x4 b = *(const f32x4*)(src + i + 4);
  v8bf o;
  #pragma unroll
  for (int j = 0; j < 4; ++j) { o[j] = (__bf16)a[j]; o[j + 4] = (__bf16)b[j]; }
  *(v8bf*)(dst + i) = o;
}

// ---------------------------------------------------------------------------
// Transpose + convert weights: Wt[n*1024+k] = (bf16)W[k*1024+n]
// ---------------------------------------------------------------------------
__global__ __launch_bounds__(256) void tcvt_kernel(const float* __restrict__ W,
                                                   __bf16* __restrict__ Wt) {
  __shared__ float t[32][33];
  int n0 = blockIdx.x * 32, k0 = blockIdx.y * 32;
  #pragma unroll
  for (int i = threadIdx.y; i < 32; i += 8)
    t[i][threadIdx.x] = W[(size_t)(k0 + i) * DMODEL + n0 + threadIdx.x];
  __syncthreads();
  #pragma unroll
  for (int i = threadIdx.y; i < 32; i += 8)
    Wt[(size_t)(n0 + i) * DMODEL + k0 + threadIdx.x] = (__bf16)t[threadIdx.x][i];
}

// ---------------------------------------------------------------------------
// bf16 WMMA GEMM:  C[M,N] = A[M,K] * Bt[N,K]^T + bias, optional scale.
// Block tile 128x64, 8 waves of 32x32 (2x2 WMMA accs). Double-buffered LDS
// staged with global_load_async_to_lds_b128; s_wait_asynccnt pipelining.
// LDS pitch 56 elems (112B): 16B aligned + conflict-free for frag loads.
// ---------------------------------------------------------------------------
#define GLD 56

template <bool OUT_BF16>
__global__ __launch_bounds__(256) void gemm_bf16_wmma(
    const __bf16* __restrict__ A, const __bf16* __restrict__ Bt,
    const float* __restrict__ bias, void* __restrict__ Cout,
    int M, int N, int K, float scale) {
  __shared__ __align__(16) __bf16 Asl[2][128 * GLD];
  __shared__ __align__(16) __bf16 Bsl[2][64 * GLD];

  const int tid = threadIdx.x;
  const int w = tid >> 5, lane = tid & 31, half = lane >> 4, l15 = lane & 15;
  const int m0 = blockIdx.y * 128, n0 = blockIdx.x * 64;
  const int wm = (w & 3) * 32, wn = (w >> 2) * 32;

  const int sr = tid >> 2;          // 0..63 staging row
  const int sc = (tid & 3) * 8;     // 0..24 staging col chunk

  v8f acc[4];
  #pragma unroll
  for (int t = 0; t < 4; ++t)
    #pragma unroll
    for (int r = 0; r < 8; ++r) acc[t][r] = 0.0f;

  // prologue: stage K-slice 0 into buffer 0 (3 async b128 per thread)
  async_copy_b128(&Asl[0][sr * GLD + sc],        &A[(size_t)(m0 + sr) * K + sc]);
  async_copy_b128(&Asl[0][(sr + 64) * GLD + sc], &A[(size_t)(m0 + sr + 64) * K + sc]);
  async_copy_b128(&Bsl[0][sr * GLD + sc],        &Bt[(size_t)(n0 + sr) * K + sc]);

  for (int kk = 0; kk < K; kk += 32) {
    const int cur = (kk >> 5) & 1;
    if (kk + 32 < K) {
      const int nxt = cur ^ 1, kn = kk + 32;
      async_copy_b128(&Asl[nxt][sr * GLD + sc],        &A[(size_t)(m0 + sr) * K + kn + sc]);
      async_copy_b128(&Asl[nxt][(sr + 64) * GLD + sc], &A[(size_t)(m0 + sr + 64) * K + kn + sc]);
      async_copy_b128(&Bsl[nxt][sr * GLD + sc],        &Bt[(size_t)(n0 + sr) * K + kn + sc]);
      wait_async3();  // in-order: current buffer's 3 ops have completed
    } else {
      wait_async0();
    }
    __syncthreads();

    const __bf16* As = Asl[cur];
    const __bf16* Bs = Bsl[cur];
    v16bf a0 = make_frag(&As[(wm + l15) * GLD + half * 8],
                         &As[(wm + l15) * GLD + half * 8 + 16]);
    v16bf a1 = make_frag(&As[(wm + 16 + l15) * GLD + half * 8],
                         &As[(wm + 16 + l15) * GLD + half * 8 + 16]);
    v16bf b0 = make_frag(&Bs[(wn + l15) * GLD + half * 8],
                         &Bs[(wn + l15) * GLD + half * 8 + 16]);
    v16bf b1 = make_frag(&Bs[(wn + 16 + l15) * GLD + half * 8],
                         &Bs[(wn + 16 + l15) * GLD + half * 8 + 16]);

    acc[0] = __builtin_amdgcn_wmma_f32_16x16x32_bf16(false, a0, false, b0, (short)0, acc[0], false, false);
    acc[1] = __builtin_amdgcn_wmma_f32_16x16x32_bf16(false, a0, false, b1, (short)0, acc[1], false, false);
    acc[2] = __builtin_amdgcn_wmma_f32_16x16x32_bf16(false, a1, false, b0, (short)0, acc[2], false, false);
    acc[3] = __builtin_amdgcn_wmma_f32_16x16x32_bf16(false, a1, false, b1, (short)0, acc[3], false, false);
    __syncthreads();
  }

  #pragma unroll
  for (int t = 0; t < 4; ++t) {
    int mrow = m0 + wm + (t >> 1) * 16 + half * 8;
    int ncol = n0 + wn + (t & 1) * 16 + l15;
    float bi = bias[ncol];
    #pragma unroll
    for (int r = 0; r < 8; ++r) {
      float val = (acc[t][r] + bi) * scale;
      size_t idx = (size_t)(mrow + r) * N + ncol;
      if (OUT_BF16) ((__bf16*)Cout)[idx] = (__bf16)val;
      else          ((float*)Cout)[idx]  = val;
    }
  }
}

// ---------------------------------------------------------------------------
// Flash attention with band+global mask. Block = (head, 128 q rows), 8 waves
// each own 16 q rows. 32-key tiles: QK^T (4 WMMA) + P*V (4 WMMA) per tile.
// K tile staged with async-to-LDS; V staged transposed through registers.
// ---------------------------------------------------------------------------
#define KLD 72   // K-tile LDS pitch (144B: aligned + conflict-free)
#define VLD 56   // V^T-tile pitch
#define PLD 56   // per-wave P scratch pitch

__global__ __launch_bounds__(256) void attn_flash_wmma(
    const __bf16* __restrict__ q, const __bf16* __restrict__ k,
    const __bf16* __restrict__ v, __bf16* __restrict__ ao) {
  __shared__ __align__(16) __bf16 Ks[32 * KLD];
  __shared__ __align__(16) __bf16 Vt[64 * VLD];
  __shared__ __align__(16) __bf16 Ps[8 * 16 * PLD];

  const int tid = threadIdx.x;
  const int w = tid >> 5, lane = tid & 31, half = lane >> 4, l15 = lane & 15;
  const int qb = blockIdx.x * 128;
  const int h = blockIdx.y, hoff = h * DH;

  // Q fragments for this wave's 16 rows (K = Dh = 64 -> two 16x32 frags)
  const __bf16* qp = q + (size_t)(qb + w * 16 + l15) * DMODEL + hoff;
  v16bf qf0 = make_frag(qp + half * 8, qp + half * 8 + 16);
  v16bf qf1 = make_frag(qp + 32 + half * 8, qp + 32 + half * 8 + 16);

  v8f O[4];
  float mrun[8], lrun[8];
  #pragma unroll
  for (int n = 0; n < 4; ++n)
    #pragma unroll
    for (int r = 0; r < 8; ++r) O[n][r] = 0.0f;
  #pragma unroll
  for (int r = 0; r < 8; ++r) { mrun[r] = -1e30f; lrun[r] = 0.0f; }

  const int skey = tid >> 3;        // 0..31 key within tile
  const int sd = (tid & 7) * 8;     // 0..56 head-dim chunk

  for (int jb = 0; jb < S_LEN / 32; ++jb) {
    const int j0 = jb * 32;
    // uniform tile-skip: band, global column tile 0, global row block 0
    bool process = (jb == 0) || (qb == 0) ||
                   ((j0 + 31 >= qb - WIN) && (j0 <= qb + 127 + WIN));
    if (!process) continue;

    // stage K tile [32 keys][64 d] async; V^T tile [64 d][32 keys] via regs
    async_copy_b128(&Ks[skey * KLD + sd],
                    &k[(size_t)(j0 + skey) * DMODEL + hoff + sd]);
    v8bf vv = *(const v8bf*)&v[(size_t)(j0 + skey) * DMODEL + hoff + sd];
    #pragma unroll
    for (int i = 0; i < 8; ++i) Vt[(sd + i) * VLD + skey] = vv[i];
    wait_async0();
    __syncthreads();

    // scores: two 16x16 tiles over K = 64
    v8f sc2[2];
    #pragma unroll
    for (int kt = 0; kt < 2; ++kt) {
      const __bf16* kb = &Ks[(kt * 16 + l15) * KLD];
      v16bf b0 = make_frag(kb + half * 8, kb + half * 8 + 16);
      v16bf b1 = make_frag(kb + 32 + half * 8, kb + 32 + half * 8 + 16);
      v8f s;
      #pragma unroll
      for (int r = 0; r < 8; ++r) s[r] = 0.0f;
      s = __builtin_amdgcn_wmma_f32_16x16x32_bf16(false, qf0, false, b0, (short)0, s, false, false);
      s = __builtin_amdgcn_wmma_f32_16x16x32_bf16(false, qf1, false, b1, (short)0, s, false, false);
      // mask: |i-j|<=WIN or i==0 or j==0
      #pragma unroll
      for (int r = 0; r < 8; ++r) {
        int i = qb + w * 16 + r + half * 8;
        int j = j0 + kt * 16 + l15;
        int d = i - j;
        bool ok = (d <= WIN && d >= -WIN) || (i == 0) || (j == 0);
        s[r] = ok ? s[r] : -1e30f;
      }
      sc2[kt] = s;
    }

    // flash update: row max over 16 lanes of each half
    float tmax[8];
    #pragma unroll
    for (int r = 0; r < 8; ++r) tmax[r] = fmaxf(sc2[0][r], sc2[1][r]);
    #pragma unroll
    for (int msk = 1; msk < 16; msk <<= 1)
      #pragma unroll
      for (int r = 0; r < 8; ++r)
        tmax[r] = fmaxf(tmax[r], __shfl_xor(tmax[r], msk, 32));

    float p0[8], p1[8], rs[8];
    #pragma unroll
    for (int r = 0; r < 8; ++r) {
      float mn = fmaxf(mrun[r], tmax[r]);
      float alpha = __expf(mrun[r] - mn);
      p0[r] = __expf(sc2[0][r] - mn);
      p1[r] = __expf(sc2[1][r] - mn);
      rs[r] = p0[r] + p1[r];
      mrun[r] = mn;
      lrun[r] *= alpha;
      #pragma unroll
      for (int n = 0; n < 4; ++n) O[n][r] *= alpha;
    }
    #pragma unroll
    for (int msk = 1; msk < 16; msk <<= 1)
      #pragma unroll
      for (int r = 0; r < 8; ++r) rs[r] += __shfl_xor(rs[r], msk, 32);
    #pragma unroll
    for (int r = 0; r < 8; ++r) lrun[r] += rs[r];

    // C-layout -> A-layout via per-wave LDS (wave-local DS is in-order)
    const int pb = w * 16 * PLD;
    #pragma unroll
    for (int r = 0; r < 8; ++r) {
      int row = r + half * 8;
      Ps[pb + row * PLD + l15]      = (__bf16)p0[r];
      Ps[pb + row * PLD + 16 + l15] = (__bf16)p1[r];
    }
    const __bf16* pp = &Ps[pb + l15 * PLD];
    v16bf pf = make_frag(pp + half * 8, pp + half * 8 + 16);

    // O += P * V
    #pragma unroll
    for (int n = 0; n < 4; ++n) {
      const __bf16* vb = &Vt[(n * 16 + l15) * VLD];
      v16bf bv = make_frag(vb + half * 8, vb + half * 8 + 16);
      O[n] = __builtin_amdgcn_wmma_f32_16x16x32_bf16(false, pf, false, bv, (short)0, O[n], false, false);
    }
    __syncthreads();
  }

  // finalize: divide by l, write bf16 [S, H*Dh]
  #pragma unroll
  for (int r = 0; r < 8; ++r) {
    float inv = 1.0f / lrun[r];
    int row = qb + w * 16 + r + half * 8;
    #pragma unroll
    for (int n = 0; n < 4; ++n)
      ao[(size_t)row * DMODEL + hoff + n * 16 + l15] = (__bf16)(O[n][r] * inv);
  }
}

// ---------------------------------------------------------------------------
extern "C" void kernel_launch(void* const* d_in, const int* in_sizes, int n_in,
                              void* d_out, int out_size, void* d_ws, size_t ws_size,
                              hipStream_t stream) {
  const float* x  = (const float*)d_in[0];
  const float* Wq = (const float*)d_in[1];
  const float* bq = (const float*)d_in[2];
  const float* Wk = (const float*)d_in[3];
  const float* bk = (const float*)d_in[4];
  const float* Wv = (const float*)d_in[5];
  const float* bv = (const float*)d_in[6];
  const float* Wo = (const float*)d_in[7];
  const float* bo = (const float*)d_in[8];
  float* out = (float*)d_out;

  char* ws = (char*)d_ws;
  const size_t XB = (size_t)S_LEN * DMODEL * 2;   // 4 MB
  const size_t WB = (size_t)DMODEL * DMODEL * 2;  // 2 MB
  __bf16* xb  = (__bf16*)(ws);
  __bf16* wqt = (__bf16*)(ws + XB);
  __bf16* wkt = (__bf16*)(ws + XB + WB);
  __bf16* wvt = (__bf16*)(ws + XB + 2 * WB);
  __bf16* wot = (__bf16*)(ws + XB + 3 * WB);
  __bf16* qb  = (__bf16*)(ws + XB + 4 * WB);
  __bf16* kb  = (__bf16*)(ws + 2 * XB + 4 * WB);
  __bf16* vb_ = (__bf16*)(ws + 3 * XB + 4 * WB);
  __bf16* ab  = (__bf16*)(ws + 4 * XB + 4 * WB);

  // 1) convert activations + transpose-convert weights
  {
    int n = S_LEN * DMODEL;
    cvt_bf16_kernel<<<n / (8 * 256), 256, 0, stream>>>(x, xb, n);
    dim3 tb(32, 8), tg(DMODEL / 32, DMODEL / 32);
    tcvt_kernel<<<tg, tb, 0, stream>>>(Wq, wqt);
    tcvt_kernel<<<tg, tb, 0, stream>>>(Wk, wkt);
    tcvt_kernel<<<tg, tb, 0, stream>>>(Wv, wvt);
    tcvt_kernel<<<tg, tb, 0, stream>>>(Wo, wot);
  }

  // 2) Q/K/V projections (scale Dh^-0.5 folded into Q)
  {
    dim3 grid(DMODEL / 64, S_LEN / 128);
    const float scale = 0.125f;  // 64^-0.5
    gemm_bf16_wmma<true><<<grid, 256, 0, stream>>>(xb, wqt, bq, qb, S_LEN, DMODEL, DMODEL, scale);
    gemm_bf16_wmma<true><<<grid, 256, 0, stream>>>(xb, wkt, bk, kb, S_LEN, DMODEL, DMODEL, 1.0f);
    gemm_bf16_wmma<true><<<grid, 256, 0, stream>>>(xb, wvt, bv, vb_, S_LEN, DMODEL, DMODEL, 1.0f);
  }

  // 3) banded+global flash attention
  {
    dim3 grid(S_LEN / 128, NHEAD);
    attn_flash_wmma<<<grid, 256, 0, stream>>>(qb, kb, vb_, ab);
  }

  // 4) output projection -> f32
  {
    dim3 grid(DMODEL / 64, S_LEN / 128);
    gemm_bf16_wmma<false><<<grid, 256, 0, stream>>>(ab, wot, bo, out, S_LEN, DMODEL, DMODEL, 1.0f);
  }
}
